// Dust3RAttention_4973572129470
// MI455X (gfx1250) — matrix operand order repaired and checked by reading, not compiled, not run on hardware
//
#include <hip/hip_runtime.h>

// ---------------------------------------------------------------------------
// Dust3R attention for MI455X (gfx1250, wave32, WMMA bf16 16x16x32)
//   B=8, N=1024, C=768, H=12, hd=64
// Pipeline:
//   K0: f32 -> bf16 pre-convert of x, w_qkv, w_proj (halves GEMM read traffic)
//   K1: qkv = x @ w_qkv     (WMMA bf16, async-to-LDS staging, double-buffered)
//   K2: RoPE(q,k) in place + V^T (elementwise)
//   K3: flash attention     (WMMA bf16, online softmax, L2-resident K/V)
//   K4: out = attn @ w_proj + b  (same GEMM, f32 out)
// ---------------------------------------------------------------------------

#define BATCH 8
#define SEQ   1024
#define CH    768
#define HEADS 12
#define HDIM  64
#define QKVC  (3 * CH)   // 2304

#ifndef __has_builtin
#define __has_builtin(x) 0
#endif

#if __has_builtin(__builtin_amdgcn_global_load_async_to_lds_b128)
#define USE_ASYNC_LDS 1
#else
#define USE_ASYNC_LDS 0
#endif

typedef __attribute__((ext_vector_type(16))) __bf16 v16bf;
typedef __attribute__((ext_vector_type(8)))  float  v8f;
typedef int v4i_vec __attribute__((vector_size(16)));   // matches builtin param

union FragBF { uint4 u[2]; v16bf v; };   // 32 bytes: 16 bf16 per lane (8 VGPRs)
union FragF  { v8f v; float f[8]; };     // 32 bytes: 8 f32 per lane (8 VGPRs)

static_assert(sizeof(FragBF) == 32, "frag size");
static_assert(sizeof(FragF)  == 32, "acc size");

__device__ __forceinline__ unsigned short f2bf(float f) {
  unsigned int u = __float_as_uint(f);
  unsigned int r = 0x7FFFu + ((u >> 16) & 1u);     // round-to-nearest-even
  return (unsigned short)((u + r) >> 16);
}
__device__ __forceinline__ float bf2f(unsigned short h) {
  return __uint_as_float(((unsigned int)h) << 16);
}

__device__ __forceinline__ v8f wmma_bf16(const v16bf& a, const v16bf& b, const v8f& c) {
  return __builtin_amdgcn_wmma_f32_16x16x32_bf16(false, a, false, b, (short)0, c,
                                                 false, false);
}

__device__ __forceinline__ float hmax16(float v) {
  v = fmaxf(v, __shfl_xor(v, 1, 32));
  v = fmaxf(v, __shfl_xor(v, 2, 32));
  v = fmaxf(v, __shfl_xor(v, 4, 32));
  v = fmaxf(v, __shfl_xor(v, 8, 32));
  return v;
}
__device__ __forceinline__ float hsum16(float v) {
  v += __shfl_xor(v, 1, 32);
  v += __shfl_xor(v, 2, 32);
  v += __shfl_xor(v, 4, 32);
  v += __shfl_xor(v, 8, 32);
  return v;
}

// async 16-byte global->LDS copy (ASYNCcnt tracked); fallback: VGPR round trip
__device__ __forceinline__ void copy16_g2l(const unsigned short* g,
                                           unsigned short* l) {
#if USE_ASYNC_LDS
  __builtin_amdgcn_global_load_async_to_lds_b128(
      (__attribute__((address_space(1))) v4i_vec*)(g),
      (__attribute__((address_space(3))) v4i_vec*)(l), 0, 0);
#else
  *(uint4*)l = *(const uint4*)g;
#endif
}

__device__ __forceinline__ void wait_async_copies() {
#if USE_ASYNC_LDS
#if __has_builtin(__builtin_amdgcn_s_wait_asynccnt)
  __builtin_amdgcn_s_wait_asynccnt(0);
#else
  asm volatile("s_wait_asynccnt 0" ::: "memory");
#endif
#endif
}

// ---------------------------------------------------------------------------
// K0: elementwise f32 -> bf16 (n must be a multiple of 4)
// ---------------------------------------------------------------------------
__global__ __launch_bounds__(256)
void f32_to_bf16_kernel(const float* __restrict__ src,
                        unsigned short* __restrict__ dst, int n4) {
  const int i = blockIdx.x * 256 + threadIdx.x;
  if (i >= n4) return;
  float4 v = *(const float4*)&src[(size_t)i * 4];
  uint2 p;
  p.x = (unsigned)f2bf(v.x) | ((unsigned)f2bf(v.y) << 16);
  p.y = (unsigned)f2bf(v.z) | ((unsigned)f2bf(v.w) << 16);
  *(uint2*)&dst[(size_t)i * 4] = p;
}

// ---------------------------------------------------------------------------
// K1 / K4: tiled WMMA GEMM, bf16 x bf16.  C[M,N] = A[M,K]*B[K,N] (+bias)
//   Block tile 128x128, K-step 32, 256 threads = 8 waves (4 x 2).
//   Double-buffered LDS; A tile staged with GLOBAL_LOAD_ASYNC_TO_LDS_B128,
//   B tile staged transposed [n][k] manually (transpose not async-copyable).
//   Fragment reads: all 16-byte LDS loads matching the ISA VGPR layouts:
//     A frag: lane m=l&15, b128 at k = 8g and 16+8g           (g = l>>4)
//     B frag: lane n=l&15, b128 at k = 8g and 16+8g
// ---------------------------------------------------------------------------
template <bool OUT_F32>
__global__ __launch_bounds__(256)
void gemm_bf16_wmma(const unsigned short* __restrict__ A,
                    const unsigned short* __restrict__ B,
                    const float* __restrict__ bias,
                    void* __restrict__ Cp,
                    int M, int Ncols, int K) {
  __shared__ __align__(16) unsigned short sA[2][128 * 40];
  __shared__ __align__(16) unsigned short sBT[2][128 * 40];

  const int tid  = threadIdx.x;
  const int lane = tid & 31;
  const int wid  = tid >> 5;
  const int ln   = lane & 15;
  const int g    = lane >> 4;
  const int wm   = (wid >> 1) * 32;
  const int wn   = (wid & 1) * 64;

  const int m0 = blockIdx.y * 128;
  const int n0 = blockIdx.x * 128;

  FragF acc[2][4];
#pragma unroll
  for (int s = 0; s < 2; ++s)
#pragma unroll
    for (int t = 0; t < 4; ++t)
#pragma unroll
      for (int j = 0; j < 8; ++j) acc[s][t].f[j] = 0.0f;

  // ---- tile stagers ----
  auto stageA = [&](int k0, int buf) {
#pragma unroll
    for (int r = 0; r < 2; ++r) {
      const int slot = tid + 256 * r;          // 0..511
      const int row  = slot >> 2;              // 0..127
      const int ch   = (slot & 3) * 8;         // 0,8,16,24  (k offset)
      copy16_g2l(&A[(size_t)(m0 + row) * K + k0 + ch],
                 &sA[buf][row * 40 + ch]);
    }
  };
  auto stageB = [&](int k0, int buf) {
    const int kk0 = tid >> 5;                  // 0..7
    const int nn  = (tid & 31) * 4;            // 0,4,..,124
#pragma unroll
    for (int r = 0; r < 4; ++r) {
      const int kk = kk0 + 8 * r;
      uint2 v = *(const uint2*)&B[(size_t)(k0 + kk) * Ncols + n0 + nn];
      sBT[buf][(nn + 0) * 40 + kk] = (unsigned short)(v.x & 0xFFFFu);
      sBT[buf][(nn + 1) * 40 + kk] = (unsigned short)(v.x >> 16);
      sBT[buf][(nn + 2) * 40 + kk] = (unsigned short)(v.y & 0xFFFFu);
      sBT[buf][(nn + 3) * 40 + kk] = (unsigned short)(v.y >> 16);
    }
  };

  // ---- prologue ----
  stageA(0, 0);
  stageB(0, 0);
  wait_async_copies();
  __syncthreads();

  int buf = 0;
  for (int k0 = 0; k0 < K; k0 += 32) {
    const bool more = (k0 + 32) < K;
    if (more) {                 // kick off next tile into the other buffer
      stageA(k0 + 32, buf ^ 1);
      stageB(k0 + 32, buf ^ 1);
    }

    FragBF fa[2], fb[4];
#pragma unroll
    for (int s = 0; s < 2; ++s) {
      const int m = wm + s * 16 + ln;
      fa[s].u[0] = *(const uint4*)&sA[buf][m * 40 + 8 * g];
      fa[s].u[1] = *(const uint4*)&sA[buf][m * 40 + 16 + 8 * g];
    }
#pragma unroll
    for (int t = 0; t < 4; ++t) {
      const int n = wn + t * 16 + ln;
      fb[t].u[0] = *(const uint4*)&sBT[buf][n * 40 + 8 * g];
      fb[t].u[1] = *(const uint4*)&sBT[buf][n * 40 + 16 + 8 * g];
    }
#pragma unroll
    for (int s = 0; s < 2; ++s)
#pragma unroll
      for (int t = 0; t < 4; ++t)
        acc[s][t].v = wmma_bf16(fa[s].v, fb[t].v, acc[s][t].v);

    if (more) wait_async_copies();
    __syncthreads();
    buf ^= 1;
  }

  // ---- epilogue: D layout lane -> (m = j + 8g, n = ln) ----
#pragma unroll
  for (int s = 0; s < 2; ++s)
#pragma unroll
    for (int t = 0; t < 4; ++t)
#pragma unroll
      for (int j = 0; j < 8; ++j) {
        const int row = m0 + wm + s * 16 + j + 8 * g;
        const int col = n0 + wn + t * 16 + ln;
        if (OUT_F32) {
          float bv = bias ? bias[col] : 0.0f;
          ((float*)Cp)[(size_t)row * Ncols + col] = acc[s][t].f[j] + bv;
        } else {
          ((unsigned short*)Cp)[(size_t)row * Ncols + col] = f2bf(acc[s][t].f[j]);
        }
      }
}

// ---------------------------------------------------------------------------
// K2: RoPE on q,k (in-place in bf16 qkv buffer) + transpose V into Vt[B,H,hd,N]
//   one thread per (b, n, h, i) with i in [0,32): handles dim pair (i, i+32)
// ---------------------------------------------------------------------------
__global__ __launch_bounds__(256)
void rope_vt_kernel(unsigned short* __restrict__ qkv,
                    unsigned short* __restrict__ vt) {
  const size_t idx = (size_t)blockIdx.x * 256 + threadIdx.x;
  const size_t total = (size_t)BATCH * SEQ * HEADS * 32;
  if (idx >= total) return;

  const int i  = (int)(idx & 31);
  const int h  = (int)((idx >> 5) % HEADS);
  const size_t bn = idx / (32 * HEADS);
  const int n  = (int)(bn % SEQ);
  const int b  = (int)(bn / SEQ);

  // inv_freq = theta^(-2i/hd); ang = n * inv_freq
  const float inv = exp2f(-((float)(2 * i) / (float)HDIM) * 13.2877123795494f);
  const float ang = (float)n * inv;
  const float c = cosf(ang), s = sinf(ang);

  const size_t base = ((size_t)(b * SEQ + n)) * QKVC + (size_t)h * HDIM;

  // q
  {
    float t1 = bf2f(qkv[base + i]);
    float t2 = bf2f(qkv[base + i + 32]);
    qkv[base + i]      = f2bf(t1 * c - t2 * s);
    qkv[base + i + 32] = f2bf(t2 * c + t1 * s);
  }
  // k
  {
    const size_t kb = base + CH;
    float t1 = bf2f(qkv[kb + i]);
    float t2 = bf2f(qkv[kb + i + 32]);
    qkv[kb + i]      = f2bf(t1 * c - t2 * s);
    qkv[kb + i + 32] = f2bf(t2 * c + t1 * s);
  }
  // v transpose -> Vt[b][h][d][n]
  {
    const size_t vb = base + 2 * CH;
    const size_t vtb = ((size_t)(b * HEADS + h)) * HDIM * SEQ;
    vt[vtb + (size_t)i * SEQ + n]        = qkv[vb + i];
    vt[vtb + (size_t)(i + 32) * SEQ + n] = qkv[vb + i + 32];
  }
}

// ---------------------------------------------------------------------------
// K3: flash attention.  Grid (N/64, H, B), block 128 = 4 waves.
//   Each wave owns 16 query rows; online softmax over 32-key blocks.
//   Q / K^T / V^T fragments are direct b128 global loads (192MB L2-resident),
//   P is transposed D->A layout through a per-wave LDS scratch.
// ---------------------------------------------------------------------------
__global__ __launch_bounds__(128)
void attn_fa_kernel(const unsigned short* __restrict__ qkv,
                    const unsigned short* __restrict__ vt,
                    unsigned short* __restrict__ ao) {
  const int b  = blockIdx.z;
  const int h  = blockIdx.y;
  const int qt = blockIdx.x;

  const int tid  = threadIdx.x;
  const int lane = tid & 31;
  const int wid  = tid >> 5;
  const int ln   = lane & 15;
  const int g    = lane >> 4;
  const int qbase = qt * 64 + wid * 16;
  const float scale = 0.125f;   // 1/sqrt(64)

  __shared__ __align__(16) unsigned short pbuf[4][16 * 32];

  // ---- Q A-fragments (16 rows x 64 hd = two 16x32 frags) ----
  FragBF fq[2];
  {
    const unsigned short* qp =
        qkv + ((size_t)(b * SEQ + qbase + ln)) * QKVC + (size_t)h * HDIM;
    fq[0].u[0] = *(const uint4*)(qp + 8 * g);
    fq[0].u[1] = *(const uint4*)(qp + 16 + 8 * g);
    fq[1].u[0] = *(const uint4*)(qp + 32 + 8 * g);
    fq[1].u[1] = *(const uint4*)(qp + 48 + 8 * g);
  }

  FragF o[4];
  float mrow[8], lrow[8];
#pragma unroll
  for (int t = 0; t < 4; ++t)
#pragma unroll
    for (int j = 0; j < 8; ++j) o[t].f[j] = 0.0f;
#pragma unroll
  for (int j = 0; j < 8; ++j) { mrow[j] = -1e30f; lrow[j] = 0.0f; }

  const unsigned short* kbasep =
      qkv + (size_t)b * SEQ * QKVC + CH + (size_t)h * HDIM;
  const unsigned short* vbasep = vt + ((size_t)(b * HEADS + h)) * HDIM * SEQ;

  for (int kb = 0; kb < SEQ; kb += 32) {
    // ---- S = Q K^T for two 16-key groups ----
    FragF sacc[2];
#pragma unroll
    for (int kg = 0; kg < 2; ++kg) {
      const int key = kb + kg * 16 + ln;
      const unsigned short* kp = kbasep + (size_t)key * QKVC;
      FragBF fk0, fk1;                    // hd chunks [0,32) and [32,64)
      fk0.u[0] = *(const uint4*)(kp + 8 * g);
      fk0.u[1] = *(const uint4*)(kp + 16 + 8 * g);
      fk1.u[0] = *(const uint4*)(kp + 32 + 8 * g);
      fk1.u[1] = *(const uint4*)(kp + 48 + 8 * g);
      v8f z = {0.f, 0.f, 0.f, 0.f, 0.f, 0.f, 0.f, 0.f};
      v8f p = wmma_bf16(fq[0].v, fk0.v, z);
      sacc[kg].v = wmma_bf16(fq[1].v, fk1.v, p);
    }

    // ---- online softmax (rows m = j + 8g, reduce over 16 n-lanes) ----
    float corr[8];
#pragma unroll
    for (int j = 0; j < 8; ++j) {
      sacc[0].f[j] *= scale;
      sacc[1].f[j] *= scale;
      float v = hmax16(fmaxf(sacc[0].f[j], sacc[1].f[j]));
      float mn = fmaxf(mrow[j], v);
      corr[j] = __expf(mrow[j] - mn);
      mrow[j] = mn;
    }
#pragma unroll
    for (int kg = 0; kg < 2; ++kg)
#pragma unroll
      for (int j = 0; j < 8; ++j)
        sacc[kg].f[j] = __expf(sacc[kg].f[j] - mrow[j]);
#pragma unroll
    for (int j = 0; j < 8; ++j) {
      float rs = hsum16(sacc[0].f[j] + sacc[1].f[j]);
      lrow[j] = lrow[j] * corr[j] + rs;
    }

    // ---- transpose P: D layout -> A layout via per-wave LDS ----
#pragma unroll
    for (int kg = 0; kg < 2; ++kg)
#pragma unroll
      for (int j = 0; j < 8; ++j)
        pbuf[wid][(j + 8 * g) * 32 + kg * 16 + ln] = f2bf(sacc[kg].f[j]);
    __asm volatile("" ::: "memory");   // same-wave DS ops are in-order

    FragBF pa;
    pa.u[0] = *(const uint4*)&pbuf[wid][ln * 32 + 8 * g];
    pa.u[1] = *(const uint4*)&pbuf[wid][ln * 32 + 16 + 8 * g];

    // ---- rescale O, accumulate P x V ----
#pragma unroll
    for (int t = 0; t < 4; ++t)
#pragma unroll
      for (int j = 0; j < 8; ++j) o[t].f[j] *= corr[j];

#pragma unroll
    for (int t = 0; t < 4; ++t) {
      const unsigned short* vp = vbasep + (size_t)(t * 16 + ln) * SEQ + kb;
      FragBF fv;
      fv.u[0] = *(const uint4*)(vp + 8 * g);
      fv.u[1] = *(const uint4*)(vp + 16 + 8 * g);
      o[t].v = wmma_bf16(pa.v, fv.v, o[t].v);
    }
  }

  // ---- normalize + store bf16 into attn_out[B,N,C] ----
#pragma unroll
  for (int t = 0; t < 4; ++t)
#pragma unroll
    for (int j = 0; j < 8; ++j) {
      const int qrow = qbase + j + 8 * g;
      const int d = t * 16 + ln;
      float val = o[t].f[j] / lrow[j];
      ao[((size_t)(b * SEQ + qrow)) * CH + (size_t)h * HDIM + d] = f2bf(val);
    }
}

// ---------------------------------------------------------------------------
// launch
// ---------------------------------------------------------------------------
extern "C" void kernel_launch(void* const* d_in, const int* in_sizes, int n_in,
                              void* d_out, int out_size, void* d_ws, size_t ws_size,
                              hipStream_t stream) {
  const float* x      = (const float*)d_in[0];   // [8,1024,768]
  const float* w_qkv  = (const float*)d_in[1];   // [768,2304]
  const float* w_proj = (const float*)d_in[2];   // [768,768]
  const float* b_proj = (const float*)d_in[3];   // [768]
  float* out          = (float*)d_out;           // [8,1024,768]

  const size_t M = (size_t)BATCH * SEQ;          // 8192
  unsigned short* p   = (unsigned short*)d_ws;
  unsigned short* xb  = p;  p += M * CH;                         // x bf16
  unsigned short* wqb = p;  p += (size_t)CH * QKVC;              // w_qkv bf16
  unsigned short* wpb = p;  p += (size_t)CH * CH;                // w_proj bf16
  unsigned short* qkv = p;  p += M * QKVC;                       // qkv bf16
  unsigned short* vt  = p;  p += (size_t)BATCH * HEADS * HDIM * SEQ; // V^T bf16
  unsigned short* ao  = p;                                       // attn out bf16

  // K0: pre-convert f32 -> bf16
  {
    const int nx = (int)(M * CH / 4);
    f32_to_bf16_kernel<<<(nx + 255) / 256, 256, 0, stream>>>(x, xb, nx);
    const int nq = CH * QKVC / 4;
    f32_to_bf16_kernel<<<(nq + 255) / 256, 256, 0, stream>>>(w_qkv, wqb, nq);
    const int np = CH * CH / 4;
    f32_to_bf16_kernel<<<(np + 255) / 256, 256, 0, stream>>>(w_proj, wpb, np);
  }
  // K1: qkv = x @ w_qkv  (bf16 out)
  {
    dim3 grid(QKVC / 128, (int)(M / 128));
    gemm_bf16_wmma<false><<<grid, 256, 0, stream>>>(
        xb, wqb, nullptr, (void*)qkv, (int)M, QKVC, CH);
  }
  // K2: RoPE(q,k) in place + V^T
  {
    const size_t total = (size_t)BATCH * SEQ * HEADS * 32;
    rope_vt_kernel<<<(unsigned)((total + 255) / 256), 256, 0, stream>>>(qkv, vt);
  }
  // K3: flash attention
  {
    dim3 grid(SEQ / 64, HEADS, BATCH);
    attn_fa_kernel<<<grid, 128, 0, stream>>>(qkv, vt, ao);
  }
  // K4: out = attn @ w_proj + b_proj  (f32 out)
  {
    dim3 grid(CH / 128, (int)(M / 128));
    gemm_bf16_wmma<true><<<grid, 256, 0, stream>>>(
        ao, wpb, b_proj, (void*)out, (int)M, CH, CH);
  }
}
